// HPTEncoder_12378095747863
// MI455X (gfx1250) — compile-verified, matrix operand order checked by reading, not run on hardware
//
#include <hip/hip_runtime.h>
#include <hip/hip_bf16.h>
#include <math.h>

// ---------------- problem constants (from reference) ----------------
#define SS   4096
#define DD   768
#define HH   12
#define DHD  64
#define LL   12
#define CC   256
#define GG   64
#define FF   3072
#define NEGB (-1.0e9f)
#define EPSV (1e-5f)

typedef __bf16 bf16_t;
typedef __attribute__((ext_vector_type(16))) __bf16 v16bf;
typedef __attribute__((ext_vector_type(8)))  float  v8f;

__device__ __forceinline__ bf16_t f2b(float x) { return (bf16_t)x; }

// CDNA5 async global->LDS copy of 16 bytes (per-lane addresses), ASYNCcnt-tracked.
// ldsoff: byte offset within workgroup LDS (low 32 bits of generic shared addr).
__device__ __forceinline__ void async_copy16(unsigned ldsoff, const void* gptr) {
  unsigned long long g = (unsigned long long)(uintptr_t)gptr;
  asm volatile("global_load_async_to_lds_b128 %0, %1, off"
               :: "v"(ldsoff), "v"(g) : "memory");
}
__device__ __forceinline__ void async_wait() {
  asm volatile("s_wait_asynccnt 0" ::: "memory");
}
__device__ __forceinline__ unsigned lds_off(const void* p) {
  return (unsigned)(uintptr_t)p;  // LDS aperture: addr[31:0] is the LDS offset
}

// D = A(16x32 bf16) * B(32x16 bf16) + C  (wave32 CDNA5 WMMA)
__device__ __forceinline__ v8f wmma_bf(v16bf a, v16bf b, v8f c) {
  return __builtin_amdgcn_wmma_f32_16x16x32_bf16(false, a, false, b, (short)0, c,
                                                 false, false);
}

// Build a 16x32 bf16 WMMA operand fragment from an LDS tile stored row-major
// with row stride `ld` (elements). Row = lane&15; CDNA5 K layout: lanes 0-15
// hold K {0..7,16..23}, lanes 16-31 hold K {8..15,24..31}; pairs contiguous.
__device__ __forceinline__ v16bf load_frag(const bf16_t* base, int ld) {
  const int lane = threadIdx.x & 31;
  const bf16_t* row = base + (lane & 15) * ld + ((lane >> 4) << 3);
  v16bf f;
#pragma unroll
  for (int e = 0; e < 4; ++e) {
    f[2 * e + 0] = row[2 * e + 0];
    f[2 * e + 1] = row[2 * e + 1];
    f[8 + 2 * e + 0] = row[16 + 2 * e + 0];
    f[8 + 2 * e + 1] = row[16 + 2 * e + 1];
  }
  return f;
}

__device__ __forceinline__ float block_sum256(float v, float* red) {
  const int tid = threadIdx.x;
  red[tid] = v;
  __syncthreads();
  for (int o = 128; o > 0; o >>= 1) {
    if (tid < o) red[tid] += red[tid + o];
    __syncthreads();
  }
  float r = red[0];
  __syncthreads();
  return r;
}

// ---------------- fp32 -> bf16 weight conversion ----------------
__global__ void k_cvt_bf16(const float* __restrict__ src, bf16_t* __restrict__ dst, int n) {
  int i = blockIdx.x * 256 + threadIdx.x;
  if (i < n) dst[i] = f2b(src[i]);
}

// ---------------- token+pos embedding + LayerNorm ----------------
__global__ __launch_bounds__(256) void k_embed(const int* __restrict__ ids,
                                               const float* __restrict__ tok,
                                               const float* __restrict__ pos,
                                               const float* __restrict__ g,
                                               const float* __restrict__ b,
                                               float* __restrict__ xf,
                                               bf16_t* __restrict__ xb) {
  __shared__ float red[256];
  const int s = blockIdx.x, tid = threadIdx.x;
  const int id = ids[s];
  float vals[3];
  float loc = 0.f;
#pragma unroll
  for (int i = 0; i < 3; ++i) {
    int d = tid + i * 256;
    float v = tok[(size_t)id * DD + d] + pos[(size_t)s * DD + d];
    vals[i] = v;
    loc += v;
  }
  const float mean = block_sum256(loc, red) * (1.0f / DD);
  loc = 0.f;
#pragma unroll
  for (int i = 0; i < 3; ++i) { float d = vals[i] - mean; loc += d * d; }
  const float var = block_sum256(loc, red) * (1.0f / DD);
  const float rstd = rsqrtf(var + EPSV);
#pragma unroll
  for (int i = 0; i < 3; ++i) {
    int d = tid + i * 256;
    float v = (vals[i] - mean) * rstd * g[d] + b[d];
    xf[(size_t)s * DD + d] = v;
    xb[(size_t)s * DD + d] = f2b(v);
  }
}

// ---------------- residual add + LayerNorm (in-place on xf, mirrors to bf16) ----
__global__ __launch_bounds__(256) void k_addln(float* __restrict__ xf,
                                               const float* __restrict__ y,
                                               const float* __restrict__ g,
                                               const float* __restrict__ b,
                                               bf16_t* __restrict__ xb) {
  __shared__ float red[256];
  const int s = blockIdx.x, tid = threadIdx.x;
  float vals[3];
  float loc = 0.f;
#pragma unroll
  for (int i = 0; i < 3; ++i) {
    int d = tid + i * 256;
    float v = xf[(size_t)s * DD + d] + y[(size_t)s * DD + d];
    vals[i] = v;
    loc += v;
  }
  const float mean = block_sum256(loc, red) * (1.0f / DD);
  loc = 0.f;
#pragma unroll
  for (int i = 0; i < 3; ++i) { float d = vals[i] - mean; loc += d * d; }
  const float var = block_sum256(loc, red) * (1.0f / DD);
  const float rstd = rsqrtf(var + EPSV);
#pragma unroll
  for (int i = 0; i < 3; ++i) {
    int d = tid + i * 256;
    float v = (vals[i] - mean) * rstd * g[d] + b[d];
    xf[(size_t)s * DD + d] = v;
    xb[(size_t)s * DD + d] = f2b(v);
  }
}

// ---------------- bf16 WMMA GEMM:  C = act((A*B + bias) * scale) ----------------
// A: MxK bf16 row-major, B: KxN bf16 row-major, block tile 128x64, 8 waves,
// each wave computes 32x32 as 2x2 WMMA tiles.  K stepped by 64 (2 WMMA
// sub-steps per stage).  A tile staged with CDNA5 async global->LDS; B tile
// transposed in-register; next tiles prefetched into L2.
__global__ __launch_bounds__(256) void k_gemm(const bf16_t* __restrict__ A,
                                              const bf16_t* __restrict__ B,
                                              const float* __restrict__ bias,
                                              float* __restrict__ Cf,
                                              bf16_t* __restrict__ Cb,
                                              int M, int N, int K,
                                              float scale, int act) {
  __shared__ bf16_t As[128 * 72];  // 128 rows x 64 K (+8 pad)
  __shared__ bf16_t Bt[64 * 72];   // 64 N-rows x 64 K (+8 pad), transposed B tile
  const int tid = threadIdx.x;
  const int wave = tid >> 5, lane = tid & 31;
  const int m0 = blockIdx.y * 128, n0 = blockIdx.x * 64;
  const int wm = (wave & 3) * 32, wn = (wave >> 2) * 32;

  v8f acc[2][2] = {};
  for (int k0 = 0; k0 < K; k0 += 64) {
    // ---- stage A tile (128x64) via async global->LDS, 4 x b128 per thread ----
#pragma unroll
    for (int i = 0; i < 4; ++i) {
      int e = (i * 256 + tid) * 8;         // element index within tile
      int r = e >> 6, c = e & 63;
      async_copy16(lds_off(&As[r * 72 + c]),
                   A + (size_t)(m0 + r) * K + k0 + c);
    }
    // ---- stage B tile transposed (K rows k0..k0+63, N cols n0..n0+63) ----
#pragma unroll
    for (int i = 0; i < 2; ++i) {
      int e = (i * 256 + tid) * 8;
      int r = e >> 6, c = e & 63;          // r = K row, c = N col
      union { uint4 u; bf16_t h[8]; } u;
      u.u = *reinterpret_cast<const uint4*>(B + (size_t)(k0 + r) * N + n0 + c);
#pragma unroll
      for (int j = 0; j < 8; ++j) Bt[(c + j) * 72 + r] = u.h[j];
    }
    // ---- prefetch next K tiles into L2 (global_prefetch_b8) ----
    if (k0 + 64 < K) {
      __builtin_prefetch(A + (size_t)(m0 + (tid >> 1)) * K + k0 + 64 + (tid & 1) * 32, 0, 1);
      __builtin_prefetch(B + (size_t)(k0 + 64 + (tid >> 2)) * N + n0 + (tid & 3) * 16, 0, 1);
    }
    async_wait();
    __syncthreads();
#pragma unroll
    for (int ks = 0; ks < 2; ++ks) {
      v16bf a0 = load_frag(&As[(wm + 0) * 72 + ks * 32], 72);
      v16bf a1 = load_frag(&As[(wm + 16) * 72 + ks * 32], 72);
      v16bf b0 = load_frag(&Bt[(wn + 0) * 72 + ks * 32], 72);
      v16bf b1 = load_frag(&Bt[(wn + 16) * 72 + ks * 32], 72);
      acc[0][0] = wmma_bf(a0, b0, acc[0][0]);
      acc[0][1] = wmma_bf(a0, b1, acc[0][1]);
      acc[1][0] = wmma_bf(a1, b0, acc[1][0]);
      acc[1][1] = wmma_bf(a1, b1, acc[1][1]);
    }
    __syncthreads();
  }
  const int half8 = (lane >> 4) << 3;
  const int cl = lane & 15;
#pragma unroll
  for (int i = 0; i < 2; ++i)
#pragma unroll
    for (int j = 0; j < 2; ++j) {
      int col = n0 + wn + j * 16 + cl;
      float bv = bias ? bias[col] : 0.f;
#pragma unroll
      for (int r = 0; r < 8; ++r) {
        int row = m0 + wm + i * 16 + half8 + r;
        float v = (acc[i][j][r] + bv) * scale;
        if (act) v = 0.5f * v * (1.0f + erff(v * 0.70710678f));  // exact GELU
        if (Cf) Cf[(size_t)row * N + col] = v;
        if (Cb) Cb[(size_t)row * N + col] = f2b(v);
      }
    }
}

// ---------------- local banded attention (flash-style, WMMA QK^T and P.V) ----
// grid = (32 half-chunks, 12 heads), 8 waves, each wave owns 16 query rows.
// Streams 26 key tiles of 32: 24 local-window tiles + 2 global tiles (keys<64).
__global__ __launch_bounds__(256) void k_attn_local(const bf16_t* __restrict__ qb,
                                                    const bf16_t* __restrict__ kb,
                                                    const bf16_t* __restrict__ vb,
                                                    const int* __restrict__ attn,
                                                    bf16_t* __restrict__ ob) {
  __shared__ bf16_t Qs[8][16 * 72];
  __shared__ bf16_t Kt[32 * 72];     // [key][d]
  __shared__ bf16_t Vt[64 * 40];     // [d][key]  (transposed for P.V B-operand)
  __shared__ float  Sb[8][16 * 34];
  __shared__ bf16_t Pb[8][16 * 40];
  __shared__ float  mst[8][16], lst[8][16], ast[8][16];
  __shared__ int    km[32];

  const int tid = threadIdx.x, wave = tid >> 5, lane = tid & 31;
  const int hb2 = blockIdx.x;                 // half-chunk 0..31
  const int head = blockIdx.y;
  const int chunk = hb2 >> 1;
  const int qabs0 = hb2 * 128 + wave * 16;    // absolute query row base
  const int qi0 = (hb2 & 1) * 128 + wave * 16;// within-chunk query row base
  const int dcol = head * DHD;
  const int kwin0 = chunk * CC - CC;

  // stage this wave's 16x64 Q tile (4 x b128 per lane)
#pragma unroll
  for (int i = 0; i < 4; ++i) {
    int e = (i * 32 + lane) * 8;
    int r = e >> 6, c = e & 63;
    union { uint4 u; bf16_t h[8]; } u;
    u.u = *reinterpret_cast<const uint4*>(qb + (size_t)(qabs0 + r) * DD + dcol + c);
#pragma unroll
    for (int j = 0; j < 8; ++j) Qs[wave][r * 72 + c + j] = u.h[j];
  }
  if (lane < 16) { mst[wave][lane] = -3.0e38f; lst[wave][lane] = 0.f; }
  __syncthreads();
  v16bf qa0 = load_frag(&Qs[wave][0], 72);
  v16bf qa1 = load_frag(&Qs[wave][32], 72);
  v8f o[4] = {};
  const int half8 = (lane >> 4) << 3;
  const int cl = lane & 15;

  for (int step = 0; step < 26; ++step) {
    const bool glob = (step >= 24);
    const int kp0 = glob ? (step - 24) * 32 : kwin0 + step * 32;
    __syncthreads();  // previous tile fully consumed
    {
      int e = tid * 8;                       // 2048 elems, one b128 per thread
      int kk = e >> 6, c = e & 63;
      int kp = kp0 + kk;
      if (kp >= 0 && kp < SS) {
        async_copy16(lds_off(&Kt[kk * 72 + c]),
                     kb + (size_t)kp * DD + dcol + c);
        union { uint4 u; bf16_t h[8]; } u;
        u.u = *reinterpret_cast<const uint4*>(vb + (size_t)kp * DD + dcol + c);
#pragma unroll
        for (int j = 0; j < 8; ++j) Vt[(c + j) * 40 + kk] = u.h[j];
      } else {
#pragma unroll
        for (int j = 0; j < 8; ++j) {
          Kt[kk * 72 + c + j] = (bf16_t)0.f;
          Vt[(c + j) * 40 + kk] = (bf16_t)0.f;
        }
      }
    }
    if (tid < 32) {
      int kp = kp0 + tid;
      int ok;
      if (glob) ok = (kp < GG) && (attn[kp] > 0);
      else      ok = (kp >= GG) && (kp < SS) && (attn[kp] > 0);
      km[tid] = ok;
    }
    async_wait();
    __syncthreads();
    // S(16x32) = Q(16x64) . K^T  via 4 WMMAs
    v16bf kf00 = load_frag(&Kt[0], 72);
    v16bf kf01 = load_frag(&Kt[32], 72);
    v16bf kf10 = load_frag(&Kt[16 * 72], 72);
    v16bf kf11 = load_frag(&Kt[16 * 72 + 32], 72);
    v8f s0 = {}, s1 = {};
    s0 = wmma_bf(qa0, kf00, s0); s0 = wmma_bf(qa1, kf01, s0);
    s1 = wmma_bf(qa0, kf10, s1); s1 = wmma_bf(qa1, kf11, s1);
#pragma unroll
    for (int r = 0; r < 8; ++r) {
      int qi = qi0 + half8 + r;
      int w0 = step * 32 + cl, w1 = w0 + 16;
      bool b0 = km[cl]      && (glob || (w0 >= qi && w0 <= qi + 2 * CC));
      bool b1 = km[16 + cl] && (glob || (w1 >= qi && w1 <= qi + 2 * CC));
      s0[r] += b0 ? 0.f : NEGB;
      s1[r] += b1 ? 0.f : NEGB;
    }
#pragma unroll
    for (int r = 0; r < 8; ++r) {
      Sb[wave][(half8 + r) * 34 + cl] = s0[r];
      Sb[wave][(half8 + r) * 34 + 16 + cl] = s1[r];
    }
    __syncthreads();
    if (lane < 16) {  // streaming softmax row update, lane = query row
      float mold = mst[wave][lane];
      float mx = mold;
#pragma unroll
      for (int c = 0; c < 32; ++c) mx = fmaxf(mx, Sb[wave][lane * 34 + c]);
      float alpha = __expf(mold - mx);
      float sum = 0.f;
#pragma unroll
      for (int c = 0; c < 32; ++c) {
        float p = __expf(Sb[wave][lane * 34 + c] - mx);
        sum += p;
        Pb[wave][lane * 40 + c] = f2b(p);
      }
      mst[wave][lane] = mx;
      lst[wave][lane] = lst[wave][lane] * alpha + sum;
      ast[wave][lane] = alpha;
    }
    __syncthreads();
#pragma unroll
    for (int r = 0; r < 8; ++r) {
      float al = ast[wave][half8 + r];
#pragma unroll
      for (int nt = 0; nt < 4; ++nt) o[nt][r] *= al;
    }
    v16bf pf = load_frag(&Pb[wave][0], 40);
#pragma unroll
    for (int nt = 0; nt < 4; ++nt) {
      v16bf vf = load_frag(&Vt[nt * 16 * 40], 40);
      o[nt] = wmma_bf(pf, vf, o[nt]);
    }
  }
  __syncthreads();
#pragma unroll
  for (int r = 0; r < 8; ++r) {
    float inv = 1.0f / fmaxf(lst[wave][half8 + r], 1e-30f);
    int row = qabs0 + half8 + r;
#pragma unroll
    for (int nt = 0; nt < 4; ++nt)
      ob[(size_t)row * DD + dcol + nt * 16 + cl] = f2b(o[nt][r] * inv);
  }
}

// ---------------- global attention (64 global query rows vs all 4096 keys) ----
// grid = 12 heads; waves 0-3 own the four 16-row query tiles (waves 4-7 mirror
// them to keep barriers uniform and do not write output).
__global__ __launch_bounds__(256) void k_attn_glob(const bf16_t* __restrict__ qb,
                                                   const bf16_t* __restrict__ kb,
                                                   const bf16_t* __restrict__ vb,
                                                   const int* __restrict__ attn,
                                                   bf16_t* __restrict__ ob) {
  __shared__ bf16_t Qs[8][16 * 72];
  __shared__ bf16_t Kt[32 * 72];
  __shared__ bf16_t Vt[64 * 40];
  __shared__ float  Sb[8][16 * 34];
  __shared__ bf16_t Pb[8][16 * 40];
  __shared__ float  mst[8][16], lst[8][16], ast[8][16];
  __shared__ int    km[32];

  const int tid = threadIdx.x, wave = tid >> 5, lane = tid & 31;
  const int head = blockIdx.x;
  const int tile = wave & 3;
  const int qrow0 = tile * 16;
  const int dcol = head * DHD;

#pragma unroll
  for (int i = 0; i < 4; ++i) {
    int e = (i * 32 + lane) * 8;
    int r = e >> 6, c = e & 63;
    union { uint4 u; bf16_t h[8]; } u;
    u.u = *reinterpret_cast<const uint4*>(qb + (size_t)(qrow0 + r) * DD + dcol + c);
#pragma unroll
    for (int j = 0; j < 8; ++j) Qs[wave][r * 72 + c + j] = u.h[j];
  }
  if (lane < 16) { mst[wave][lane] = -3.0e38f; lst[wave][lane] = 0.f; }
  __syncthreads();
  v16bf qa0 = load_frag(&Qs[wave][0], 72);
  v16bf qa1 = load_frag(&Qs[wave][32], 72);
  v8f o[4] = {};
  const int half8 = (lane >> 4) << 3;
  const int cl = lane & 15;

  for (int step = 0; step < SS / 32; ++step) {
    const int kp0 = step * 32;
    __syncthreads();
    {
      int e = tid * 8;
      int kk = e >> 6, c = e & 63;
      int kp = kp0 + kk;
      async_copy16(lds_off(&Kt[kk * 72 + c]),
                   kb + (size_t)kp * DD + dcol + c);
      union { uint4 u; bf16_t h[8]; } u;
      u.u = *reinterpret_cast<const uint4*>(vb + (size_t)kp * DD + dcol + c);
#pragma unroll
      for (int j = 0; j < 8; ++j) Vt[(c + j) * 40 + kk] = u.h[j];
    }
    if (tid < 32) km[tid] = (attn[kp0 + tid] > 0);
    async_wait();
    __syncthreads();
    v16bf kf00 = load_frag(&Kt[0], 72);
    v16bf kf01 = load_frag(&Kt[32], 72);
    v16bf kf10 = load_frag(&Kt[16 * 72], 72);
    v16bf kf11 = load_frag(&Kt[16 * 72 + 32], 72);
    v8f s0 = {}, s1 = {};
    s0 = wmma_bf(qa0, kf00, s0); s0 = wmma_bf(qa1, kf01, s0);
    s1 = wmma_bf(qa0, kf10, s1); s1 = wmma_bf(qa1, kf11, s1);
#pragma unroll
    for (int r = 0; r < 8; ++r) {
      s0[r] += km[cl] ? 0.f : NEGB;
      s1[r] += km[16 + cl] ? 0.f : NEGB;
    }
#pragma unroll
    for (int r = 0; r < 8; ++r) {
      Sb[wave][(half8 + r) * 34 + cl] = s0[r];
      Sb[wave][(half8 + r) * 34 + 16 + cl] = s1[r];
    }
    __syncthreads();
    if (lane < 16) {
      float mold = mst[wave][lane];
      float mx = mold;
#pragma unroll
      for (int c = 0; c < 32; ++c) mx = fmaxf(mx, Sb[wave][lane * 34 + c]);
      float alpha = __expf(mold - mx);
      float sum = 0.f;
#pragma unroll
      for (int c = 0; c < 32; ++c) {
        float p = __expf(Sb[wave][lane * 34 + c] - mx);
        sum += p;
        Pb[wave][lane * 40 + c] = f2b(p);
      }
      mst[wave][lane] = mx;
      lst[wave][lane] = lst[wave][lane] * alpha + sum;
      ast[wave][lane] = alpha;
    }
    __syncthreads();
#pragma unroll
    for (int r = 0; r < 8; ++r) {
      float al = ast[wave][half8 + r];
#pragma unroll
      for (int nt = 0; nt < 4; ++nt) o[nt][r] *= al;
    }
    v16bf pf = load_frag(&Pb[wave][0], 40);
#pragma unroll
    for (int nt = 0; nt < 4; ++nt) {
      v16bf vf = load_frag(&Vt[nt * 16 * 40], 40);
      o[nt] = wmma_bf(pf, vf, o[nt]);
    }
  }
  __syncthreads();
  if (wave < 4) {
#pragma unroll
    for (int r = 0; r < 8; ++r) {
      float inv = 1.0f / fmaxf(lst[wave][half8 + r], 1e-30f);
      int row = qrow0 + half8 + r;
#pragma unroll
      for (int nt = 0; nt < 4; ++nt)
        ob[(size_t)row * DD + dcol + nt * 16 + cl] = f2b(o[nt][r] * inv);
    }
  }
}

// ---------------- weighted BCE loss ----------------
__global__ void k_zero(float* a) { if (threadIdx.x < 8) a[threadIdx.x] = 0.f; }

__global__ void k_count(const int* __restrict__ me, const int* __restrict__ ms,
                        float* __restrict__ a) {
  int s = blockIdx.x * 256 + threadIdx.x;
  if (s >= SS) return;
  float ne = 0, pe = 0, ns = 0, ps = 0;
#pragma unroll
  for (int j = 0; j < 3; ++j) {
    int m = me[s * 3 + j];
    if (m != 0) { ne += 1.f; pe += (m + 1) * 0.5f; }
  }
#pragma unroll
  for (int j = 0; j < 2; ++j) {
    int m = ms[s * 2 + j];
    if (m != 0) { ns += 1.f; ps += (m + 1) * 0.5f; }
  }
  atomicAdd(&a[0], ne); atomicAdd(&a[1], pe);
  atomicAdd(&a[2], ns); atomicAdd(&a[3], ps);
}

__global__ void k_bce(const float* __restrict__ x,
                      const float* __restrict__ aw, const float* __restrict__ ab,
                      const float* __restrict__ sw, const float* __restrict__ sb,
                      const int* __restrict__ me, const int* __restrict__ ms,
                      float* __restrict__ a) {
  int s = blockIdx.x * 256 + threadIdx.x;
  if (s >= SS) return;
  float pa[3] = {ab[0], ab[1], ab[2]};
  float pp[2] = {sb[0], sb[1]};
  const float* h = x + (size_t)s * DD;
  for (int d = 0; d < DD; ++d) {
    float hv = h[d];
    pa[0] += hv * aw[d * 3 + 0];
    pa[1] += hv * aw[d * 3 + 1];
    pa[2] += hv * aw[d * 3 + 2];
    pp[0] += hv * sw[d * 2 + 0];
    pp[1] += hv * sw[d * 2 + 1];
  }
  const float n_e = a[0], p_e = a[1], n_s = a[2], p_s = a[3];
  float le = 0.f, ls = 0.f;
#pragma unroll
  for (int j = 0; j < 3; ++j) {
    int m = me[s * 3 + j];
    if (m != 0) {
      float pred = 1.f / (1.f + __expf(-pa[j]));
      float tgt = (m + 1) * 0.5f;
      float w = (m == 1) ? (n_e / p_e * 0.5f) : (n_e / (n_e - p_e) * 0.5f);
      float p = fminf(fmaxf(pred, 1e-7f), 1.f - 1e-7f);
      le += w * (-(tgt * __logf(p) + (1.f - tgt) * log1pf(-p)));
    }
  }
#pragma unroll
  for (int j = 0; j < 2; ++j) {
    int m = ms[s * 2 + j];
    if (m != 0) {
      float pred = 1.f / (1.f + __expf(-pp[j]));
      float tgt = (m + 1) * 0.5f;
      float w = (m == 1) ? (n_s / p_s * 0.5f) : (n_s / (n_s - p_s) * 0.5f);
      float p = fminf(fmaxf(pred, 1e-7f), 1.f - 1e-7f);
      ls += w * (-(tgt * __logf(p) + (1.f - tgt) * log1pf(-p)));
    }
  }
  atomicAdd(&a[4], le);
  atomicAdd(&a[5], ls);
}

__global__ void k_final(const float* __restrict__ a, float* __restrict__ out) {
  out[0] = 0.7f * (a[4] / a[0]) + 0.3f * (a[5] / a[2]);
}

// ---------------- host launcher ----------------
static inline size_t alup(size_t x) { return (x + 255) & ~(size_t)255; }

extern "C" void kernel_launch(void* const* d_in, const int* in_sizes, int n_in,
                              void* d_out, int out_size, void* d_ws, size_t ws_size,
                              hipStream_t stream) {
  (void)in_sizes; (void)n_in; (void)out_size; (void)ws_size;
  const int*   ids   = (const int*)d_in[0];
  const int*   attn  = (const int*)d_in[2];
  const int*   mle   = (const int*)d_in[4];   // mask_label_HTMLelements (S,3)
  const int*   mas   = (const int*)d_in[5];   // mask_answer_span (S,2)
  const float* tok   = (const float*)d_in[6];
  const float* pos   = (const float*)d_in[7];
  const float* elg   = (const float*)d_in[8];
  const float* elb   = (const float*)d_in[9];
  const float* wq  = (const float*)d_in[10]; const float* bq  = (const float*)d_in[11];
  const float* wk  = (const float*)d_in[12]; const float* bk  = (const float*)d_in[13];
  const float* wv  = (const float*)d_in[14]; const float* bv  = (const float*)d_in[15];
  const float* wqg = (const float*)d_in[16]; const float* bqg = (const float*)d_in[17];
  const float* wkg = (const float*)d_in[18]; const float* bkg = (const float*)d_in[19];
  const float* wvg = (const float*)d_in[20]; const float* bvg = (const float*)d_in[21];
  const float* wo  = (const float*)d_in[22]; const float* bo  = (const float*)d_in[23];
  const float* l1g = (const float*)d_in[24]; const float* l1b = (const float*)d_in[25];
  const float* w1  = (const float*)d_in[26]; const float* b1  = (const float*)d_in[27];
  const float* w2  = (const float*)d_in[28]; const float* b2  = (const float*)d_in[29];
  const float* l2g = (const float*)d_in[30]; const float* l2b = (const float*)d_in[31];
  const float* answ = (const float*)d_in[32]; const float* ansb = (const float*)d_in[33];
  const float* spnw = (const float*)d_in[34]; const float* spnb = (const float*)d_in[35];

  char* w = (char*)d_ws;
  size_t off = 0;
  auto carve = [&](size_t bytes) { void* p = w + off; off = alup(off + bytes); return p; };
  float*  xf  = (float*)carve((size_t)SS * DD * 4);
  bf16_t* xb  = (bf16_t*)carve((size_t)SS * DD * 2);
  bf16_t* qb  = (bf16_t*)carve((size_t)SS * DD * 2);
  bf16_t* kb2 = (bf16_t*)carve((size_t)SS * DD * 2);
  bf16_t* vb2 = (bf16_t*)carve((size_t)SS * DD * 2);
  bf16_t* qgb = (bf16_t*)carve((size_t)SS * DD * 2);
  bf16_t* kgb = (bf16_t*)carve((size_t)SS * DD * 2);
  bf16_t* vgb = (bf16_t*)carve((size_t)SS * DD * 2);
  bf16_t* ab2 = (bf16_t*)carve((size_t)SS * DD * 2);
  float*  tf  = (float*)carve((size_t)SS * DD * 4);
  bf16_t* hb  = (bf16_t*)carve((size_t)SS * FF * 2);
  bf16_t* wbf = (bf16_t*)carve((size_t)DD * FF * 2);
  float*  acc = (float*)carve(64);

  const dim3 gP(DD / 64, SS / 128);   // 12 x 32 blocks, N = 768
  const dim3 gF(FF / 64, SS / 128);   // 48 x 32 blocks, N = 3072
  const float qscale = 0.125f;        // 1/sqrt(DH)
  const int cvDD = (DD * DD + 255) / 256;
  const int cvDF = (DD * FF + 255) / 256;

  k_embed<<<SS, 256, 0, stream>>>(ids, tok, pos, elg, elb, xf, xb);

  for (int l = 0; l < LL; ++l) {
    const size_t oDD = (size_t)l * DD * DD, oDF = (size_t)l * DD * FF;
    // Q,K,V projections
    k_cvt_bf16<<<cvDD, 256, 0, stream>>>(wq + oDD, wbf, DD * DD);
    k_gemm<<<gP, 256, 0, stream>>>(xb, wbf, bq + l * DD, nullptr, qb, SS, DD, DD, qscale, 0);
    k_cvt_bf16<<<cvDD, 256, 0, stream>>>(wk + oDD, wbf, DD * DD);
    k_gemm<<<gP, 256, 0, stream>>>(xb, wbf, bk + l * DD, nullptr, kb2, SS, DD, DD, 1.0f, 0);
    k_cvt_bf16<<<cvDD, 256, 0, stream>>>(wv + oDD, wbf, DD * DD);
    k_gemm<<<gP, 256, 0, stream>>>(xb, wbf, bv + l * DD, nullptr, vb2, SS, DD, DD, 1.0f, 0);
    // global-attention projections
    k_cvt_bf16<<<cvDD, 256, 0, stream>>>(wqg + oDD, wbf, DD * DD);
    k_gemm<<<gP, 256, 0, stream>>>(xb, wbf, bqg + l * DD, nullptr, qgb, SS, DD, DD, qscale, 0);
    k_cvt_bf16<<<cvDD, 256, 0, stream>>>(wkg + oDD, wbf, DD * DD);
    k_gemm<<<gP, 256, 0, stream>>>(xb, wbf, bkg + l * DD, nullptr, kgb, SS, DD, DD, 1.0f, 0);
    k_cvt_bf16<<<cvDD, 256, 0, stream>>>(wvg + oDD, wbf, DD * DD);
    k_gemm<<<gP, 256, 0, stream>>>(xb, wbf, bvg + l * DD, nullptr, vgb, SS, DD, DD, 1.0f, 0);
    // attention
    k_attn_local<<<dim3(32, HH), 256, 0, stream>>>(qb, kb2, vb2, attn, ab2);
    k_attn_glob<<<HH, 256, 0, stream>>>(qgb, kgb, vgb, attn, ab2);
    // output projection + residual LN
    k_cvt_bf16<<<cvDD, 256, 0, stream>>>(wo + oDD, wbf, DD * DD);
    k_gemm<<<gP, 256, 0, stream>>>(ab2, wbf, bo + l * DD, tf, nullptr, SS, DD, DD, 1.0f, 0);
    k_addln<<<SS, 256, 0, stream>>>(xf, tf, l1g + l * DD, l1b + l * DD, xb);
    // FFN
    k_cvt_bf16<<<cvDF, 256, 0, stream>>>(w1 + oDF, wbf, DD * FF);
    k_gemm<<<gF, 256, 0, stream>>>(xb, wbf, b1 + l * FF, nullptr, hb, SS, FF, DD, 1.0f, 1);
    k_cvt_bf16<<<cvDF, 256, 0, stream>>>(w2 + oDF, wbf, FF * DD);
    k_gemm<<<gP, 256, 0, stream>>>(hb, wbf, b2 + l * DD, tf, nullptr, SS, DD, FF, 1.0f, 0);
    k_addln<<<SS, 256, 0, stream>>>(xf, tf, l2g + l * DD, l2b + l * DD, xb);
  }

  k_zero<<<1, 32, 0, stream>>>(acc);
  k_count<<<SS / 256, 256, 0, stream>>>(mle, mas, acc);
  k_bce<<<SS / 256, 256, 0, stream>>>(xf, answ, ansb, spnw, spnb, mle, mas, acc);
  k_final<<<1, 1, 0, stream>>>(acc, (float*)d_out);
}
